// GeoSparseResidualBlock_53180285059157
// MI455X (gfx1250) — compile-verified, matrix operand order checked by reading, not run on hardware
//
#include <hip/hip_runtime.h>
#include <hip/hip_bf16.h>
#include <stdint.h>

// ---------------- problem constants ----------------
constexpr int Fdim  = 1024;   // feature dim
constexpr int Kspr  = 32;     // sparse fan-in per output
constexpr int Nrows = 2048;   // batch rows
constexpr int TM    = 32;     // rows per block (two 16-row WMMA sub-tiles)
constexpr float EPS = 1e-6f;

// ---------------- WMMA types ----------------
typedef __attribute__((ext_vector_type(16))) __bf16 v16bf;
typedef __attribute__((ext_vector_type(8)))  __bf16 v8bf;
typedef __attribute__((ext_vector_type(8)))  float  v8f;

// float <-> bf16 bits (round-to-nearest-even), no scalar __bf16 arithmetic
__device__ __forceinline__ unsigned short f2bf_bits(float f) {
    uint32_t u = __builtin_bit_cast(uint32_t, f);
    u += 0x7FFFu + ((u >> 16) & 1u);
    return (unsigned short)(u >> 16);
}
__device__ __forceinline__ float bf2f_bits(unsigned short h) {
    uint32_t u = ((uint32_t)h) << 16;
    return __builtin_bit_cast(float, u);
}

// Load one WMMA 16-bit operand fragment for this lane: two contiguous
// 8-element (16-byte) chunks at columns [c0, c0+8) and [c0+16, c0+24).
// Matches the CDNA5 16-bit A/B 16x32 VGPR layout when the caller passes
// c0 = kk + 8*(lane>>4) and row = lane&15.
__device__ __forceinline__ v16bf ld_frag(const unsigned short* p) {
    v8bf lo = *(const v8bf*)(p);
    v8bf hi = *(const v8bf*)(p + 16);
    v16bf r;
#pragma unroll
    for (int j = 0; j < 8; ++j) { r[j] = lo[j]; r[j + 8] = hi[j]; }
    return r;
}

__device__ __forceinline__ v8f wmma_bf16(v16bf a, v16bf b, v8f c) {
    // (neg_a, A, neg_b, B, c_mod, C, reuse_a, reuse_b)
    return __builtin_amdgcn_wmma_f32_16x16x32_bf16(false, a, false, b,
                                                   (short)0, c, false, false);
}

// ---------------- kernel 1: zero densified weights ----------------
__global__ void gsr_zero_kernel(uint32_t* __restrict__ p, int ndwords) {
    int t = blockIdx.x * blockDim.x + threadIdx.x;
    if (t < ndwords) p[t] = 0u;
}

// ---------------- kernel 2: scatter sparse weights into dense bf16 ----------
// One thread per (layer, output-row): serial 32-element scatter -> no races,
// duplicate indices accumulate (matches einsum semantics).
__global__ void gsr_densify_kernel(const int* __restrict__ idx,
                                   const float* __restrict__ w1,
                                   const float* __restrict__ w2,
                                   unsigned short* __restrict__ Wd) {
    int t = blockIdx.x * blockDim.x + threadIdx.x;
    if (t >= 2 * Fdim) return;
    int which = t >> 10;            // 0 -> layer1, 1 -> layer2
    int o     = t & (Fdim - 1);
    const float* w = which ? w2 : w1;
    unsigned short* W = Wd + (size_t)which * Fdim * Fdim + (size_t)o * Fdim;
    for (int k = 0; k < Kspr; ++k) {
        int i = idx[o * Kspr + k];
        float cur = bf2f_bits(W[i]);
        W[i] = f2bf_bits(cur + w[o * Kspr + k]);
    }
}

// ---------------- kernel 3: fused rmsnorm -> GEMM1 -> sine -> GEMM2 -> residual
__global__ __launch_bounds__(256)
void gsr_fused_kernel(const float* __restrict__ x,
                      const float* __restrict__ norm_w,
                      const float* __restrict__ b1,
                      const float* __restrict__ amp,
                      const float* __restrict__ freq,
                      const float* __restrict__ decay,
                      const float* __restrict__ b2,
                      const float* __restrict__ alpha,
                      const unsigned short* __restrict__ Wd1,
                      const unsigned short* __restrict__ Wd2,
                      float* __restrict__ out) {
    __shared__ unsigned short hS[TM][Fdim];   // rmsnorm'd activations (bf16)
    __shared__ unsigned short sS[TM][Fdim];   // sine-layer activations (bf16)
    __shared__ float redS[TM][8];
    __shared__ float scaleS[TM];

    const int tid  = threadIdx.x;
    const int row0 = blockIdx.x * TM;

    // ---- phase 0: RMSNorm 32 rows into hS ----
    {
        const int r = tid >> 3;   // 0..31  (8 threads per row)
        const int c = tid & 7;
        const float* xr = x + (size_t)(row0 + r) * Fdim;
        float acc = 0.f;
        for (int j = c; j < Fdim; j += 8) { float v = xr[j]; acc += v * v; }
        redS[r][c] = acc;
    }
    __syncthreads();
    if (tid < TM) {
        float s = 0.f;
#pragma unroll
        for (int j = 0; j < 8; ++j) s += redS[tid][j];
        scaleS[tid] = rsqrtf(s * (1.0f / Fdim) + EPS);
    }
    __syncthreads();
    for (int e = tid; e < TM * Fdim; e += 256) {
        int r = e >> 10, c = e & (Fdim - 1);
        hS[r][c] = f2bf_bits(x[(size_t)(row0 + r) * Fdim + c] * scaleS[r] * norm_w[c]);
    }
    __syncthreads();

    const int wave = tid >> 5;
    const int lane = tid & 31;
    const int nl   = lane & 15;          // row of A / col of B within 16-tile
    const int kh   = lane >> 4;          // k-half select
    const int kb   = kh * 8;             // column sub-offset for this lane

    // Each wave owns 8 contiguous o-tiles: [wave*8, wave*8+8), in 2 passes of 4.

    // ---- phase 1: z = hS * Wd1^T ; sS = amp*sin(freq*z)*exp(-decay*z^2) ----
#pragma unroll 1
    for (int j0 = 0; j0 < 8; j0 += 4) {
        const int obase0 = (wave * 8 + j0) * 16;
        const unsigned short* Bb = Wd1 + (size_t)(obase0 + nl) * Fdim + kb;
        v8f acc[8];
#pragma unroll
        for (int j = 0; j < 8; ++j) acc[j] = (v8f){};
#pragma unroll 2
        for (int kk = 0; kk < Fdim; kk += 32) {
            v16bf a0 = ld_frag(&hS[nl][kk + kb]);
            v16bf a1 = ld_frag(&hS[nl + 16][kk + kb]);
#pragma unroll
            for (int j = 0; j < 4; ++j) {
                v16bf b = ld_frag(Bb + (size_t)j * 16 * Fdim + kk);
                acc[2 * j]     = wmma_bf16(a0, b, acc[2 * j]);
                acc[2 * j + 1] = wmma_bf16(a1, b, acc[2 * j + 1]);
            }
        }
#pragma unroll
        for (int j = 0; j < 4; ++j) {
            const int oc = obase0 + j * 16 + nl;
            const float bb = b1[oc], am = amp[oc], fr = freq[oc], dc = decay[oc];
#pragma unroll
            for (int r = 0; r < 8; ++r) {
                int m = r + kh * 8;       // C/D layout: lanes16-31 hold M+8
                float z0 = acc[2 * j][r] + bb;
                sS[m][oc] = f2bf_bits(am * __sinf(fr * z0) * __expf(-dc * z0 * z0));
                float z1 = acc[2 * j + 1][r] + bb;
                sS[m + 16][oc] = f2bf_bits(am * __sinf(fr * z1) * __expf(-dc * z1 * z1));
            }
        }
    }
    __syncthreads();

    // ---- phase 2: h2 = sS * Wd2^T ; out = x + alpha*(h2 + b2) ----
    const float al = alpha[0];
#pragma unroll 1
    for (int j0 = 0; j0 < 8; j0 += 4) {
        const int obase0 = (wave * 8 + j0) * 16;
        const unsigned short* Bb = Wd2 + (size_t)(obase0 + nl) * Fdim + kb;
        v8f acc[8];
#pragma unroll
        for (int j = 0; j < 8; ++j) acc[j] = (v8f){};
#pragma unroll 2
        for (int kk = 0; kk < Fdim; kk += 32) {
            v16bf a0 = ld_frag(&sS[nl][kk + kb]);
            v16bf a1 = ld_frag(&sS[nl + 16][kk + kb]);
#pragma unroll
            for (int j = 0; j < 4; ++j) {
                v16bf b = ld_frag(Bb + (size_t)j * 16 * Fdim + kk);
                acc[2 * j]     = wmma_bf16(a0, b, acc[2 * j]);
                acc[2 * j + 1] = wmma_bf16(a1, b, acc[2 * j + 1]);
            }
        }
#pragma unroll
        for (int j = 0; j < 4; ++j) {
            const int oc = obase0 + j * 16 + nl;
            const float bb = b2[oc];
#pragma unroll
            for (int r = 0; r < 8; ++r) {
                int m0 = row0 + r + kh * 8;
                out[(size_t)m0 * Fdim + oc] =
                    x[(size_t)m0 * Fdim + oc] + al * (acc[2 * j][r] + bb);
                int m1 = m0 + 16;
                out[(size_t)m1 * Fdim + oc] =
                    x[(size_t)m1 * Fdim + oc] + al * (acc[2 * j + 1][r] + bb);
            }
        }
    }
}

// ---------------- launcher ----------------
extern "C" void kernel_launch(void* const* d_in, const int* in_sizes, int n_in,
                              void* d_out, int out_size, void* d_ws, size_t ws_size,
                              hipStream_t stream) {
    const float* x     = (const float*)d_in[0];
    const int*   idx   = (const int*)d_in[1];
    const float* normw = (const float*)d_in[2];
    const float* w1    = (const float*)d_in[3];
    const float* b1    = (const float*)d_in[4];
    const float* amp   = (const float*)d_in[5];
    const float* freq  = (const float*)d_in[6];
    const float* decay = (const float*)d_in[7];
    const float* w2    = (const float*)d_in[8];
    const float* b2    = (const float*)d_in[9];
    const float* alpha = (const float*)d_in[10];
    float* out = (float*)d_out;

    // workspace: two densified bf16 weight matrices, 2 * 1024*1024 * 2B = 4 MB
    unsigned short* Wd = (unsigned short*)d_ws;

    const int ndwords = Fdim * Fdim;  // 4 MB of zeros as dwords
    gsr_zero_kernel<<<(ndwords + 255) / 256, 256, 0, stream>>>((uint32_t*)d_ws, ndwords);
    gsr_densify_kernel<<<(2 * Fdim + 255) / 256, 256, 0, stream>>>(idx, w1, w2, Wd);
    gsr_fused_kernel<<<Nrows / TM, 256, 0, stream>>>(
        x, normw, b1, amp, freq, decay, b2, alpha,
        Wd, Wd + (size_t)Fdim * Fdim, out);
}
